// IGNNConv_65738769433062
// MI455X (gfx1250) — matrix operand order repaired and checked by reading, not compiled, not run on hardware
//
#include <hip/hip_runtime.h>
#include <hip/hip_bf16.h>

typedef __attribute__((ext_vector_type(16))) __bf16 v16bf;
typedef __attribute__((ext_vector_type(8)))  float  v8f;
typedef __attribute__((ext_vector_type(4)))  int    v4i;
typedef __attribute__((address_space(1))) v4i* gv4i_p;   // global int4*
typedef __attribute__((address_space(3))) v4i* sv4i_p;   // LDS int4*

#define HF 128   // hidden feats
#define NT 8     // 128/16 column tiles per row block
#define TILE_ELEMS (NT * 32 * 16)  // bf16 elems per 32-K B tile (8 KB)

#if __has_builtin(__builtin_amdgcn_global_load_async_to_lds_b128)
#define HAVE_ASYNC_LDS 1
#else
#define HAVE_ASYNC_LDS 0
#endif

// ---------------------------------------------------------------------------
// Small helper kernels
// ---------------------------------------------------------------------------
__global__ void zero_u32(unsigned* __restrict__ p, int n) {
  int i = blockIdx.x * blockDim.x + threadIdx.x;
  if (i < n) p[i] = 0u;
}

__global__ void count_deg(const int* __restrict__ rows, unsigned* __restrict__ deg, int nedges) {
  int e = blockIdx.x * blockDim.x + threadIdx.x;
  if (e < nedges) atomicAdd(&deg[rows[e]], 1u);
}

__global__ void calc_dinv(const unsigned* __restrict__ deg, float* __restrict__ dinv, int n) {
  int i = blockIdx.x * blockDim.x + threadIdx.x;
  if (i < n) dinv[i] = __frsqrt_rn((float)deg[i] + 1.0f);  // +1 = self loop
}

// One wave per node: 1/max(||x||,1e-12) over 512 fp32 feats.
__global__ __launch_bounds__(256) void row_invnorm(const float* __restrict__ x,
                                                   float* __restrict__ rn, int n) {
  int node = blockIdx.x * 8 + (threadIdx.x >> 5);
  if (node >= n) return;
  int lane = threadIdx.x & 31;
  const float4* p = (const float4*)(x + (size_t)node * 512);
  float ss = 0.0f;
#pragma unroll
  for (int i = 0; i < 4; ++i) {
    float4 v = p[lane + 32 * i];
    ss += v.x * v.x + v.y * v.y + v.z * v.z + v.w * v.w;
  }
#pragma unroll
  for (int off = 16; off >= 1; off >>= 1) ss += __shfl_xor(ss, off, 32);
  if (lane == 0) rn[node] = 1.0f / fmaxf(sqrtf(ss), 1e-12f);
}

// ---------------------------------------------------------------------------
// Pack W [K x 128] fp32 (row-major, B[k][n]) into per-lane WMMA B-fragment
// order (bf16).  Fragment (kt, t): lane L holds 16 contiguous bf16:
//   j -> B[kt*32 + (L>>4)*16 + j][t*16 + (L&15)]
// so VGPR v of the v16bf holds the K-pair (kbase+2v, kbase+2v+1) (contiguous k order, as the A fragment uses).
// ---------------------------------------------------------------------------
__global__ void pack_b(const float* __restrict__ W, __bf16* __restrict__ P, int K) {
  int idx = blockIdx.x * blockDim.x + threadIdx.x;  // one thread per (kt,t,lane)
  int total = (K >> 5) * NT * 32;
  if (idx >= total) return;
  int lane = idx & 31;
  int t    = (idx >> 5) & (NT - 1);
  int kt   = idx >> 8;
  int kbase = kt * 32 + (lane >> 4) * 16;
  int n     = t * 16 + (lane & 15);
  __bf16* dst = P + (size_t)idx * 16;
#pragma unroll
  for (int j = 0; j < 16; ++j)
    dst[j] = (__bf16)W[(size_t)(kbase + j) * HF + n];
}

// ---------------------------------------------------------------------------
// Fused GEMM + bias + LayerNorm + ReLU.
//   out[m, 0:128] = relu(LN(rowscale[m]*A[m, 0:K] @ B + bias))
// One wave computes a 16x128 tile (8 f32 WMMA accumulators). The shared
// 32-K-slice of B (8 KB) is staged into LDS double-buffered by the whole
// workgroup via GLOBAL_LOAD_ASYNC_TO_LDS_B128, so the 8 waves read B from
// LDS (ds_load_b128) instead of 8x re-fetching it from L2. All 8 B fragments
// are loaded into distinct registers before the WMMA burst so up to 16
// ds_load_b128 stay in flight (no per-WMMA s_wait_dscnt 0 serialization).
// A is loaded fp32 (optionally row-scaled) and packed to bf16 in-register in
// the exact A-fragment layout. LayerNorm is reduced with shfl_xor (masks 8..1
// stay within the 16-lane group holding one row of the C layout).
// ---------------------------------------------------------------------------
__global__ __launch_bounds__(256) void gemm_ln_relu(
    const float* __restrict__ A, int lda, int K,
    const float* __restrict__ rowscale,            // nullptr -> 1.0
    const __bf16* __restrict__ Bpack,
    const float* __restrict__ bias, const float* __restrict__ gamma,
    const float* __restrict__ beta,
    float* __restrict__ out, int ldo, int nrows) {
  __shared__ alignas(64) __bf16 Bsh[2][TILE_ELEMS];

  const int tid  = threadIdx.x;
  const int lane = tid & 31;
  const int wave = tid >> 5;
  const int rowBase = (blockIdx.x * 8 + wave) * 16;   // may exceed nrows on tail;
                                                      // all waves stay for barriers
  const int half = lane >> 4;
  const int l16  = lane & 15;
  int m  = rowBase + l16;
  int mc = m < nrows ? m : nrows - 1;      // clamp A loads on the tail tile
  float rs = rowscale ? rowscale[mc] : 1.0f;

  float bt[NT], gt[NT], et[NT];
#pragma unroll
  for (int t = 0; t < NT; ++t) {
    bt[t] = bias[t * 16 + l16];
    gt[t] = gamma[t * 16 + l16];
    et[t] = beta[t * 16 + l16];
  }

  // Cooperative stage of one 8 KB B tile: 512 x 16B chunks, 2 per thread.
  auto stage = [&](int kt, int buf) {
    const __bf16* src = Bpack + (size_t)kt * TILE_ELEMS;
#pragma unroll
    for (int c = 0; c < 2; ++c) {
      int chunk = tid + c * 256;
#if HAVE_ASYNC_LDS
      __builtin_amdgcn_global_load_async_to_lds_b128(
          (gv4i_p)(src + chunk * 8),          // addrspacecast -> global int4*
          (sv4i_p)(&Bsh[buf][chunk * 8]),     // addrspacecast -> LDS int4*
          0, 0);
#else
      *(float4*)(&Bsh[buf][chunk * 8]) = *(const float4*)(src + chunk * 8);
#endif
    }
  };

  v8f acc[NT] = {};
  const int nkt = K >> 5;
  const float* arow = A + (size_t)mc * lda;

  stage(0, 0);
  for (int kt = 0; kt < nkt; ++kt) {
    const int cur = kt & 1;
#if HAVE_ASYNC_LDS
#if __has_builtin(__builtin_amdgcn_s_wait_asynccnt)
    __builtin_amdgcn_s_wait_asynccnt(0);
#else
    asm volatile("s_wait_asynccnt 0x0" ::: "memory");
#endif
#endif
    __syncthreads();          // tile `cur` staged; prev reads of `cur^1` drained
    if (kt + 1 < nkt) stage(kt + 1, cur ^ 1);

    // A fragment: VGPR v holds K-pair k0 = half*16 + 2*v
    v16bf afrag;
#pragma unroll
    for (int v = 0; v < 8; ++v) {
      int k0 = half * 16 + 2 * v;
      float2 p = *(const float2*)(arow + kt * 32 + k0);
      afrag[2 * v]     = (__bf16)(p.x * rs);
      afrag[2 * v + 1] = (__bf16)(p.y * rs);
    }
    // Burst-load all 8 B fragments (distinct regs), then burst the WMMAs.
    v16bf bfrag[NT];
#pragma unroll
    for (int t = 0; t < NT; ++t)
      bfrag[t] = *(const v16bf*)(&Bsh[cur][((t * 32) + lane) * 16]);
#pragma unroll
    for (int t = 0; t < NT; ++t)
      acc[t] = __builtin_amdgcn_wmma_f32_16x16x32_bf16(
          false, afrag, false, bfrag[t], (short)0, acc[t], false, false);
  }

  // Epilogue: bias + LayerNorm(128) + ReLU, per row m = rowBase + v + 8*half
#pragma unroll
  for (int v = 0; v < 8; ++v) {
    float s = 0.0f, s2 = 0.0f;
#pragma unroll
    for (int t = 0; t < NT; ++t) {
      float c = acc[t][v] + bt[t];
      acc[t][v] = c;
      s += c;
      s2 += c * c;
    }
#pragma unroll
    for (int off = 8; off >= 1; off >>= 1) {  // stays within 16-lane group
      s  += __shfl_xor(s, off, 32);
      s2 += __shfl_xor(s2, off, 32);
    }
    float mean = s * (1.0f / 128.0f);
    float var  = s2 * (1.0f / 128.0f) - mean * mean;
    float rstd = __frsqrt_rn(var + 1e-5f);
    int row = rowBase + v + 8 * half;
    if (row < nrows) {
#pragma unroll
      for (int t = 0; t < NT; ++t) {
        float y = (acc[t][v] - mean) * rstd * gt[t] + et[t];
        out[(size_t)row * ldo + t * 16 + l16] = fmaxf(y, 0.0f);
      }
    }
  }
}

// ---------------------------------------------------------------------------
// SpMM  out = D^-1/2 (A+I) D^-1/2 z  over 128 feats.
// init: out[n] = dinv[n]^2 * z[n]     (self loop, also zero-fills)
// edges: out[row] += dinv[row]*dinv[col] * z[col]   (f32 HW atomics, L2-resident)
// ---------------------------------------------------------------------------
__global__ void spmm_init(const float* __restrict__ z, const float* __restrict__ dinv,
                          float* __restrict__ outz, int n) {
  int idx = blockIdx.x * blockDim.x + threadIdx.x;  // one per float4 (n*32 total)
  if (idx >= n * 32) return;
  int node = idx >> 5;
  float d = dinv[node];
  float w = d * d;
  float4 zv = ((const float4*)z)[idx];
  float4 o = make_float4(w * zv.x, w * zv.y, w * zv.z, w * zv.w);
  ((float4*)outz)[idx] = o;
}

__global__ __launch_bounds__(256) void spmm_edges(
    const int* __restrict__ rows, const int* __restrict__ cols,
    const float* __restrict__ dinv, const float* __restrict__ z,
    float* __restrict__ outz, int nedges) {
  int e = blockIdx.x * 8 + (threadIdx.x >> 5);  // one wave per edge
  if (e >= nedges) return;
  int lane = threadIdx.x & 31;
  int r = rows[e], c = cols[e];
  float w = dinv[r] * dinv[c];
  float4 zv = ((const float4*)(z + (size_t)c * HF))[lane];
  float* op = outz + (size_t)r * HF + lane * 4;
  unsafeAtomicAdd(op + 0, w * zv.x);
  unsafeAtomicAdd(op + 1, w * zv.y);
  unsafeAtomicAdd(op + 2, w * zv.z);
  unsafeAtomicAdd(op + 3, w * zv.w);
}

// ---------------------------------------------------------------------------
// Host orchestration
// ---------------------------------------------------------------------------
extern "C" void kernel_launch(void* const* d_in, const int* in_sizes, int n_in,
                              void* d_out, int out_size, void* d_ws, size_t ws_size,
                              hipStream_t stream) {
  const int*   ei     = (const int*)d_in[0];
  const float* feats  = (const float*)d_in[1];
  const float* W_uni  = (const float*)d_in[2];
  const float* b_uni  = (const float*)d_in[3];
  const float* g_uni  = (const float*)d_in[4];
  const float* be_uni = (const float*)d_in[5];
  const float* W_ind  = (const float*)d_in[6];
  const float* b_ind  = (const float*)d_in[7];
  const float* g_ind  = (const float*)d_in[8];
  const float* be_ind = (const float*)d_in[9];
  const float* W_rel  = (const float*)d_in[10];
  const float* b_rel  = (const float*)d_in[11];
  const float* g_rel  = (const float*)d_in[12];
  const float* be_rel = (const float*)d_in[13];

  const int E = in_sizes[0] / 2;
  const int N = in_sizes[1] / 512;
  const int* rows = ei;
  const int* cols = ei + E;

  char* ws = (char*)d_ws;
  size_t off = 0;
  auto carve = [&](size_t bytes) -> void* {
    void* p = ws + off;
    off = (off + bytes + 511) & ~(size_t)511;
    return p;
  };
  unsigned* deg     = (unsigned*)carve((size_t)N * 4);
  float*    dinv    = (float*)carve((size_t)N * 4);
  float*    rn      = (float*)carve((size_t)N * 4);
  __bf16*   packUni = (__bf16*)carve((size_t)16 * TILE_ELEMS * 2);
  __bf16*   packInd = (__bf16*)carve((size_t)4 * 4 * TILE_ELEMS * 2);
  __bf16*   packRel = (__bf16*)carve((size_t)16 * TILE_ELEMS * 2);
  float*    zA      = (float*)carve((size_t)N * HF * 4);
  float*    zB      = (float*)carve((size_t)N * HF * 4);
  float*    cat     = (float*)carve((size_t)N * 4 * HF * 4);
  float*    outp    = (float*)d_out;

  const int tb = 256;
  // Graph normalization
  zero_u32<<<(N + tb - 1) / tb, tb, 0, stream>>>(deg, N);
  count_deg<<<(E + tb - 1) / tb, tb, 0, stream>>>(rows, deg, E);
  calc_dinv<<<(N + tb - 1) / tb, tb, 0, stream>>>(deg, dinv, N);
  // Feature row norms (fused into GEMM1's A load as a row scale)
  row_invnorm<<<(N + 7) / 8, tb, 0, stream>>>(feats, rn, N);
  // Weight packing (bf16, WMMA fragment order)
  pack_b<<<(4096 + tb - 1) / tb, tb, 0, stream>>>(W_uni, packUni, 512);
  for (int i = 0; i < 4; ++i)
    pack_b<<<(1024 + tb - 1) / tb, tb, 0, stream>>>(
        W_ind + (size_t)i * HF * HF, packInd + (size_t)i * 4 * TILE_ELEMS, 128);
  pack_b<<<(4096 + tb - 1) / tb, tb, 0, stream>>>(W_rel, packRel, 512);

  const int gGemm = (N + 127) / 128;
  // Unified embedding: h = MLP(normalize(x) @ W_uni)
  gemm_ln_relu<<<gGemm, tb, 0, stream>>>(feats, 512, 512, rn, packUni,
                                         b_uni, g_uni, be_uni, zA, HF, N);
  // Inceptive hops
  float* zin = zA;
  float* zout = zB;
  for (int hop = 0; hop < 4; ++hop) {
    gemm_ln_relu<<<gGemm, tb, 0, stream>>>(
        zin, HF, HF, nullptr, packInd + (size_t)hop * 4 * TILE_ELEMS,
        b_ind + hop * HF, g_ind + hop * HF, be_ind + hop * HF,
        cat + hop * HF, 4 * HF, N);
    if (hop < 3) {
      spmm_init<<<((size_t)N * 32 + tb - 1) / tb, tb, 0, stream>>>(zin, dinv, zout, N);
      spmm_edges<<<(E + 7) / 8, tb, 0, stream>>>(rows, cols, dinv, zin, zout, E);
      float* t = zin; zin = zout; zout = t;
    }
  }
  // Relation network on concat
  gemm_ln_relu<<<gGemm, tb, 0, stream>>>(cat, 4 * HF, 512, nullptr, packRel,
                                         b_rel, g_rel, be_rel, outp, HF, N);
}
